// NeuralRadianceField_61100204753698
// MI455X (gfx1250) — matrix operand order repaired and checked by reading, not compiled, not run on hardware
//
#include <hip/hip_runtime.h>

typedef __attribute__((ext_vector_type(16))) _Float16 v16h;
typedef __attribute__((ext_vector_type(8)))  float    v8f;

#define MT   32    // points (rows) per workgroup
#define AST  328   // LDS activation row stride in f16 (320 cols + 8 pad)

union AF { v16h v; float4 f4[2]; };

// ---- A fragment: 16x32 f16 tile from LDS (row-major, stride AST) ----
// ISA layout: lanes 0-15 -> M=lane, K chunks {kb..kb+7, kb+16..kb+23}
//             lanes16-31 -> M=lane-16, K chunks {kb+8..kb+15, kb+24..kb+31}
__device__ inline v16h ld_a(const _Float16* X, int rb, int kb, int lane) {
  int r = rb + (lane & 15);
  int k = kb + ((lane >> 4) << 3);
  const _Float16* p = X + r * AST + k;
  AF t;
  t.f4[0] = *(const float4*)(p);
  t.f4[1] = *(const float4*)(p + 16);
  return t.v;
}

// ---- B fragment: 32x16 f16 tile from global W^T (N x Kpad row-major) ----
// ISA layout: lanes 0-15 -> col n=nb+lane, K chunks {kb..kb+7, kb+16..kb+23} ; lanes 16-31 -> K chunks {kb+8..kb+15, kb+24..kb+31}
__device__ inline v16h ld_b(const _Float16* WT, int Kpad, int nb, int kb, int lane) {
  int nn = nb + (lane & 15);
  int k  = kb + ((lane >> 4) << 3);
  const _Float16* p = WT + (size_t)nn * Kpad + k;
  AF t;
  t.f4[0] = *(const float4*)(p);
  t.f4[1] = *(const float4*)(p + 16);
  return t.v;
}

// ---- One dense layer: Y(32 x 16*8*CT) = act(X(32 x KPAD) @ W + b) ----
// 8 waves; each wave owns CT 16-col tiles starting at wave*CT*16.
template<int KPAD, int CT, bool RELU>
__device__ inline void mlp_layer(const _Float16* X, _Float16* Y,
                                 const _Float16* WT, const float* bias,
                                 int wave, int lane) {
  const int nbase = wave * (CT * 16);
  v8f acc[2][CT] = {};
  for (int kb = 0; kb < KPAD; kb += 32) {
    v16h a0 = ld_a(X, 0,  kb, lane);
    v16h a1 = ld_a(X, 16, kb, lane);
#pragma unroll
    for (int ct = 0; ct < CT; ++ct) {
      v16h b = ld_b(WT, KPAD, nbase + ct * 16, kb, lane);
      acc[0][ct] = __builtin_amdgcn_wmma_f32_16x16x32_f16(
          false, a0, false, b, (short)0, acc[0][ct], false, false);
      acc[1][ct] = __builtin_amdgcn_wmma_f32_16x16x32_f16(
          false, a1, false, b, (short)0, acc[1][ct], false, false);
    }
  }
  // epilogue: C layout -> VGPR i holds row i (+8 for lanes>=16), col = lane&15
#pragma unroll
  for (int ct = 0; ct < CT; ++ct) {
    int col = nbase + ct * 16 + (lane & 15);
    float bv = bias[col];
#pragma unroll
    for (int rt = 0; rt < 2; ++rt) {
      int rb = rt * 16 + ((lane >> 4) * 8);
#pragma unroll
      for (int i = 0; i < 8; ++i) {
        float v = acc[rt][ct][i] + bv;
        if (RELU) v = fmaxf(v, 0.f);
        Y[(rb + i) * AST + col] = (_Float16)v;
      }
    }
  }
}

// ---- weight convert: out[n*Kpad + k] = f16(W[k*N + n]), zero-pad k>=K ----
__global__ void cvt_wT(const float* __restrict__ W, _Float16* __restrict__ out,
                       int K, int N, int Kpad) {
  int idx = blockIdx.x * 256 + threadIdx.x;
  if (idx >= N * Kpad) return;
  int n = idx / Kpad;
  int k = idx - n * Kpad;
  out[idx] = (k < K) ? (_Float16)W[(size_t)k * N + n] : (_Float16)0.f;
}

struct NerfParams {
  const float* pts;
  const _Float16* wT[8];
  const float* b[8];
  const _Float16* wfT;  const float* bf;
  const _Float16* wc1T; const float* bc1;
  const float* wd;  const float* bd;
  const float* wc2; const float* bc2;
  float* out_d;  // P floats
  float* out_c;  // P*3 floats
};

__global__ __launch_bounds__(256, 1)
void nerf_mlp(NerfParams prm, int Ptotal) {
  __shared__ __align__(16) _Float16 A[MT * AST];
  __shared__ __align__(16) _Float16 B[MT * AST];
  const int tid = threadIdx.x;
  const int wave = tid >> 5, lane = tid & 31;
  const int base = blockIdx.x * MT;

  // ---- harmonic embedding into cols 256..319 of BOTH buffers ----
  {
    int r  = tid >> 3;               // point row 0..31
    int c0 = (tid & 7) * 8;          // 8 cols per thread
    int pidx = base + r; if (pidx >= Ptotal) pidx = Ptotal - 1;
    const float* pp = prm.pts + (size_t)pidx * 3;
    float xv0 = pp[0], xv1 = pp[1], xv2 = pp[2];
#pragma unroll
    for (int j = 0; j < 8; ++j) {
      int c = c0 + j;
      float v;
      if (c < 30) {
        int a = c / 10, f = c - 10 * a;
        float xx = (a == 0) ? xv0 : ((a == 1) ? xv1 : xv2);
        v = __sinf(xx * (float)(1 << f));
      } else if (c < 60) {
        int cc = c - 30; int a = cc / 10, f = cc - 10 * a;
        float xx = (a == 0) ? xv0 : ((a == 1) ? xv1 : xv2);
        v = __cosf(xx * (float)(1 << f));
      } else if (c == 60) v = xv0;
      else if (c == 61)   v = xv1;
      else if (c == 62)   v = xv2;
      else                v = 0.f;   // pad col 63 (K 63->64, 319->320)
      A[r * AST + 256 + c] = (_Float16)v;
      B[r * AST + 256 + c] = (_Float16)v;
    }
  }
  __syncthreads();

  // ---- trunk MLP (ping-pong A<->B; pe lives at cols 256..319 of both) ----
  mlp_layer< 64, 2, true>(A + 256, B, prm.wT[0], prm.b[0], wave, lane); __syncthreads();
  mlp_layer<256, 2, true>(B, A, prm.wT[1], prm.b[1], wave, lane); __syncthreads();
  mlp_layer<256, 2, true>(A, B, prm.wT[2], prm.b[2], wave, lane); __syncthreads();
  mlp_layer<256, 2, true>(B, A, prm.wT[3], prm.b[3], wave, lane); __syncthreads();
  mlp_layer<256, 2, true>(A, B, prm.wT[4], prm.b[4], wave, lane); __syncthreads();
  mlp_layer<320, 2, true>(B, A, prm.wT[5], prm.b[5], wave, lane); __syncthreads(); // skip-concat
  mlp_layer<256, 2, true>(A, B, prm.wT[6], prm.b[6], wave, lane); __syncthreads();
  mlp_layer<256, 2, true>(B, A, prm.wT[7], prm.b[7], wave, lane); __syncthreads();
  // final x now in A cols 0..255

  // ---- density head: relu(x @ wd + bd), spread over all threads ----
  {
    int p = tid >> 3, j = tid & 7;
    const _Float16* xr = A + p * AST;
    float s = 0.f;
#pragma unroll
    for (int kk = 0; kk < 32; ++kk) {
      int k = j * 32 + kk;
      s += (float)xr[k] * prm.wd[k];
    }
    s += __shfl_xor(s, 1, 32);
    s += __shfl_xor(s, 2, 32);
    s += __shfl_xor(s, 4, 32);
    if (j == 0 && base + p < Ptotal)
      prm.out_d[base + p] = fmaxf(s + prm.bd[0], 0.f);
  }

  // ---- feature layer (no relu): feat = x @ wf + bf -> B ----
  mlp_layer<256, 2, false>(A, B, prm.wfT, prm.bf, wave, lane); __syncthreads();

  // ---- wc1: h = relu(feat @ wc1 + bc1), 256->128 -> A cols 0..127 ----
  mlp_layer<256, 1, true>(B, A, prm.wc1T, prm.bc1, wave, lane); __syncthreads();

  // ---- color head: sigmoid(h @ wc2 + bc2), 128->3 ----
  {
    int p = tid >> 3, j = tid & 7;
    const _Float16* hr = A + p * AST;
    float s0 = 0.f, s1 = 0.f, s2 = 0.f;
#pragma unroll
    for (int kk = 0; kk < 16; ++kk) {
      int k = j * 16 + kk;
      float hv = (float)hr[k];
      s0 += hv * prm.wc2[k * 3 + 0];
      s1 += hv * prm.wc2[k * 3 + 1];
      s2 += hv * prm.wc2[k * 3 + 2];
    }
    s0 += __shfl_xor(s0, 1, 32); s0 += __shfl_xor(s0, 2, 32); s0 += __shfl_xor(s0, 4, 32);
    s1 += __shfl_xor(s1, 1, 32); s1 += __shfl_xor(s1, 2, 32); s1 += __shfl_xor(s1, 4, 32);
    s2 += __shfl_xor(s2, 1, 32); s2 += __shfl_xor(s2, 2, 32); s2 += __shfl_xor(s2, 4, 32);
    if (j == 0 && base + p < Ptotal) {
      float* oc = prm.out_c + (size_t)(base + p) * 3;
      oc[0] = 1.f / (1.f + __expf(-(s0 + prm.bc2[0])));
      oc[1] = 1.f / (1.f + __expf(-(s1 + prm.bc2[1])));
      oc[2] = 1.f / (1.f + __expf(-(s2 + prm.bc2[2])));
    }
  }
}

extern "C" void kernel_launch(void* const* d_in, const int* in_sizes, int n_in,
                              void* d_out, int out_size, void* d_ws, size_t ws_size,
                              hipStream_t stream) {
  const float* pts = (const float*)d_in[0];
  const int P = in_sizes[0] / 3;

  _Float16* ws = (_Float16*)d_ws;

  // f16-transposed weight layout in d_ws (offsets in f16 elements)
  size_t off = 0;
  size_t o_w[8], o_wf, o_wc1;
  const int Kpad_l[8] = {64, 256, 256, 256, 256, 320, 256, 256};
  const int K_l[8]    = {63, 256, 256, 256, 256, 319, 256, 256};
  for (int l = 0; l < 8; ++l) { o_w[l] = off; off += (size_t)256 * Kpad_l[l]; }
  o_wf  = off; off += (size_t)256 * 256;
  o_wc1 = off; off += (size_t)128 * 256;
  // total = 589,824 f16 (~1.2 MB) -- fits standard scratch

  // convert weights (tiny; all on stream, graph-capture safe)
  for (int l = 0; l < 8; ++l) {
    const float* W = (const float*)d_in[1 + 2 * l];
    int tot = 256 * Kpad_l[l];
    cvt_wT<<<(tot + 255) / 256, 256, 0, stream>>>(W, ws + o_w[l], K_l[l], 256, Kpad_l[l]);
  }
  {
    int tot = 256 * 256;
    cvt_wT<<<(tot + 255) / 256, 256, 0, stream>>>((const float*)d_in[19], ws + o_wf, 256, 256, 256);
  }
  {
    int tot = 128 * 256;
    cvt_wT<<<(tot + 255) / 256, 256, 0, stream>>>((const float*)d_in[21], ws + o_wc1, 256, 128, 256);
  }

  NerfParams prm;
  prm.pts = pts;
  for (int l = 0; l < 8; ++l) {
    prm.wT[l] = ws + o_w[l];
    prm.b[l]  = (const float*)d_in[2 + 2 * l];
  }
  prm.wd   = (const float*)d_in[17]; prm.bd  = (const float*)d_in[18];
  prm.wfT  = ws + o_wf;              prm.bf  = (const float*)d_in[20];
  prm.wc1T = ws + o_wc1;             prm.bc1 = (const float*)d_in[22];
  prm.wc2  = (const float*)d_in[23]; prm.bc2 = (const float*)d_in[24];
  prm.out_d = (float*)d_out;
  prm.out_c = (float*)d_out + P;

  int nblocks = (P + MT - 1) / MT;
  nerf_mlp<<<nblocks, 256, 0, stream>>>(prm, P);
}
